// NucleiGNN_21895743275105
// MI455X (gfx1250) — matrix operand order repaired and checked by reading, not compiled, not run on hardware
//
#include <hip/hip_runtime.h>
#include <hip/hip_bf16.h>
#include <math.h>

typedef __attribute__((ext_vector_type(16))) __bf16 v16bf;
typedef __attribute__((ext_vector_type(8)))  __bf16 v8bf;
typedef __attribute__((ext_vector_type(8)))  float  v8f;

#define BN 64      // nodes per molecule
#define FF 128     // feature dim
#define DH 64      // per-head dim
#define NL 3       // layers
#define MATSZ (FF*FF)

// ---- LDS layout (bytes) ------------------------------------------------
#define OFF_H      0        // float  [64][128]  32768
#define OFF_HN     32768    // bf16   [64][128]  16384
#define OFF_Q      49152    // bf16   [64][128]  16384   (reused as msg)
#define OFF_K      65536    // bf16   [64][128]  16384
#define OFF_VT     81920    // bf16   [128][64]  16384   (v transposed)
#define OFF_ATTN   98304    // bf16   [2][64][64] 16384
#define OFF_SCORES 114688   // float  [2][64][64] 32768
#define OFF_COORD  147456   // float  [64][3]      768
#define OFF_MASK   148224   // float  [64]         256
#define SMEM_BYTES 148480

static __device__ __forceinline__ v16bf cat8(v8bf lo, v8bf hi) {
  return __builtin_shufflevector(lo, hi, 0,1,2,3,4,5,6,7,8,9,10,11,12,13,14,15);
}

// A-fragment for 16x32 bf16 tile: lane (row = m0 + lane&15), two 16B chunks:
// elems 0..15 -> K = kbase + (lane>>4)*16 + e
static __device__ __forceinline__ v16bf load_a(const __bf16* p) {
  v8bf lo = *(const v8bf*)p;
  v8bf hi = *(const v8bf*)(p + 8);
  return cat8(lo, hi);
}

static __device__ __forceinline__ v8f vzero() {
  v8f z = {0.f,0.f,0.f,0.f,0.f,0.f,0.f,0.f};
  return z;
}

// GEMM: out[64x128] = A[64x128](bf16, row-major LDS) x Wsw[128x128]
// (bf16 global, pre-swizzled B-fragments). 8 waves: wave owns
// rows (wave>>1)*16.. and N-tiles (wave&1)*4..+3. K pipelined (double buffer).
// mode 0: out_bf[row*128+col] = bf16(acc)
// mode 1: out_bf[col*64+row]  = bf16(acc)           (transposed store, for v)
// mode 2: h_s[row*128+col]   += acc                 (attention residual)
// mode 3: h_s[row*128+col]   += tanh(acc+bias[col]) (FFN residual)
static __device__ void gemm_a_w(const __bf16* __restrict__ A,
                                const __bf16* __restrict__ Wsw,
                                float* h_s, __bf16* out_bf,
                                const float* __restrict__ bias,
                                int mode, int wave, int lane) {
  const int mt   = wave >> 1;
  const int ntb  = (wave & 1) * 4;
  const int half = lane >> 4;
  const int arow = mt * 16 + (lane & 15);

  v8f acc[4];
#pragma unroll
  for (int j = 0; j < 4; ++j) acc[j] = vzero();

  // prologue: fragments for kt = 0
  v16bf a = load_a(A + arow * FF + half * 16);
  v16bf b[4];
#pragma unroll
  for (int j = 0; j < 4; ++j)
    b[j] = *(const v16bf*)(Wsw + (((ntb + j) * 4 + 0) * 32 + lane) * 16);

#pragma unroll
  for (int kt = 0; kt < 4; ++kt) {
    v16bf an = a;
    v16bf bn[4];
    if (kt < 3) {
      an = load_a(A + arow * FF + (kt + 1) * 32 + half * 16);
#pragma unroll
      for (int j = 0; j < 4; ++j)
        bn[j] = *(const v16bf*)(Wsw + (((ntb + j) * 4 + kt + 1) * 32 + lane) * 16);
    }
#pragma unroll
    for (int j = 0; j < 4; ++j)
      acc[j] = __builtin_amdgcn_wmma_f32_16x16x32_bf16(
          false, a, false, b[j], (short)0, acc[j], false, false);
    if (kt < 3) {
      a = an;
#pragma unroll
      for (int j = 0; j < 4; ++j) b[j] = bn[j];
    }
  }

  const int c = lane & 15;
  const int rbase = mt * 16 + half * 8;
#pragma unroll
  for (int j = 0; j < 4; ++j) {
    const int col = (ntb + j) * 16 + c;
#pragma unroll
    for (int r = 0; r < 8; ++r) {
      const int row = rbase + r;
      float v = acc[j][r];
      if (mode == 0)      out_bf[row * FF + col] = (__bf16)v;
      else if (mode == 1) out_bf[col * BN + row] = (__bf16)v;
      else if (mode == 2) h_s[row * FF + col] += v;
      else                h_s[row * FF + col] += tanhf(v + bias[col]);
    }
  }
}

// per-row layernorm of fp32 h -> bf16 dst (threads 0..63, one row each)
static __device__ void ln_rows_bf16(const float* h_s, __bf16* dst, int tid) {
  if (tid < BN) {
    const float* r = h_s + tid * FF;
    float m = 0.f;
    for (int i = 0; i < FF; ++i) m += r[i];
    m *= (1.f / FF);
    float v = 0.f;
    for (int i = 0; i < FF; ++i) { float d = r[i] - m; v += d * d; }
    v *= (1.f / FF);
    float inv = rsqrtf(v + 1e-5f);
    __bf16* o = dst + tid * FF;
    for (int i = 0; i < FF; ++i) o[i] = (__bf16)((r[i] - m) * inv);
  }
}

// ---- pre-pass: fp32 weights -> bf16, pre-swizzled into WMMA B layout ----
__global__ void __launch_bounds__(256)
convert_weights(const float* __restrict__ Wq, const float* __restrict__ Wk,
                const float* __restrict__ Wv, const float* __restrict__ Wo,
                const float* __restrict__ Wf, __bf16* __restrict__ dst) {
  int gid = blockIdx.x * blockDim.x + threadIdx.x;
  if (gid >= 15 * MATSZ) return;
  int mi   = gid / MATSZ;        // matrix index: group*3 + layer
  int r    = gid % MATSZ;
  int nt   = r >> 11;            // N tile (0..7)
  int r2   = r & 2047;
  int kt   = r2 >> 9;            // K tile (0..3)
  int r3   = r2 & 511;
  int lane = r3 >> 4;            // 0..31
  int e    = r3 & 15;            // elem within lane's 32B
  int n    = nt * 16 + (lane & 15);
  int k    = kt * 32 + (lane >> 4) * 16 + e;
  int group = mi / 3, l = mi % 3;
  const float* src = (group == 0) ? Wq : (group == 1) ? Wk :
                     (group == 2) ? Wv : (group == 3) ? Wo : Wf;
  dst[gid] = (__bf16)src[l * MATSZ + k * FF + n];
}

// ---- main: one workgroup (8 waves) per batch element --------------------
__global__ void __launch_bounds__(256)
gnn_forward(const float* __restrict__ coords_g,
            const int* __restrict__ species_g,
            const unsigned char* __restrict__ mask_g,
            const float* __restrict__ embed_g,
            const float* __restrict__ We_g,
            const float* __restrict__ bf_g,
            const __bf16* __restrict__ wsw,
            float* __restrict__ out_g) {
  extern __shared__ char smem[];
  float*  h_s     = (float*)(smem + OFF_H);
  __bf16* hn_bf   = (__bf16*)(smem + OFF_HN);
  __bf16* q_bf    = (__bf16*)(smem + OFF_Q);
  __bf16* k_bf    = (__bf16*)(smem + OFF_K);
  __bf16* vt_bf   = (__bf16*)(smem + OFF_VT);
  __bf16* attn_bf = (__bf16*)(smem + OFF_ATTN);
  float*  scores  = (float*)(smem + OFF_SCORES);
  float*  coord_s = (float*)(smem + OFF_COORD);
  float*  mask_s  = (float*)(smem + OFF_MASK);

  const int b    = blockIdx.x;
  const int tid  = threadIdx.x;
  const int wave = tid >> 5;
  const int lane = tid & 31;

  if (tid < BN) mask_s[tid] = mask_g[b * BN + tid] ? 1.f : 0.f;
  for (int i = tid; i < BN * 3; i += 256) coord_s[i] = coords_g[b * BN * 3 + i];
  __syncthreads();

  // h = embed[species-1] * mask
  for (int idx = tid; idx < BN * FF; idx += 256) {
    int n = idx >> 7, f = idx & 127;
    int sp = species_g[b * BN + n] - 1;
    h_s[idx] = mask_s[n] * embed_g[sp * FF + f];
  }
  __syncthreads();

  for (int l = 0; l < NL; ++l) {
    const __bf16* wq = wsw + (0 * NL + l) * MATSZ;
    const __bf16* wk = wsw + (1 * NL + l) * MATSZ;
    const __bf16* wv = wsw + (2 * NL + l) * MATSZ;
    const __bf16* wo = wsw + (3 * NL + l) * MATSZ;
    const __bf16* wf = wsw + (4 * NL + l) * MATSZ;

    ln_rows_bf16(h_s, hn_bf, tid);
    __syncthreads();

    gemm_a_w(hn_bf, wq, nullptr, q_bf, nullptr, 0, wave, lane);
    gemm_a_w(hn_bf, wk, nullptr, k_bf, nullptr, 0, wave, lane);
    gemm_a_w(hn_bf, wv, nullptr, vt_bf, nullptr, 1, wave, lane);
    __syncthreads();

    // raw logits = q . k^T per head (scale applied in softmax pass)
    {
      const int half = lane >> 4;
#pragma unroll
      for (int i = 0; i < 4; ++i) {
        int t = wave * 4 + i;
        int hh = t >> 4, rem = t & 15, mt = rem >> 2, ntm = rem & 3;
        // preload both K-step fragment pairs, then issue both WMMAs
        v16bf a0 = load_a(q_bf + (mt * 16 + (lane & 15)) * FF + hh * DH + half * 16);
        v16bf b0 = *(const v16bf*)(k_bf + (ntm * 16 + (lane & 15)) * FF + hh * DH + half * 16);
        v16bf a1 = load_a(q_bf + (mt * 16 + (lane & 15)) * FF + hh * DH + 32 + half * 16);
        v16bf b1 = *(const v16bf*)(k_bf + (ntm * 16 + (lane & 15)) * FF + hh * DH + 32 + half * 16);
        v8f acc = vzero();
        acc = __builtin_amdgcn_wmma_f32_16x16x32_bf16(false, a0, false, b0, (short)0, acc, false, false);
        acc = __builtin_amdgcn_wmma_f32_16x16x32_bf16(false, a1, false, b1, (short)0, acc, false, false);
        const int c = lane & 15;
#pragma unroll
        for (int r = 0; r < 8; ++r)
          scores[hh * 4096 + (mt * 16 + half * 8 + r) * BN + ntm * 16 + c] = acc[r];
      }
    }
    __syncthreads();

    // softmax with on-the-fly edge featurization + We bias + mask
    if (tid < 128) {
      int hh = tid >> 6, n = tid & 63;
      float we[7];
#pragma unroll
      for (int e = 0; e < 7; ++e) we[e] = We_g[l * 14 + e * 2 + hh];
      float cx = coord_s[n * 3 + 0], cy = coord_s[n * 3 + 1], cz = coord_s[n * 3 + 2];
      float mn = mask_s[n];
      float* srow = scores + hh * 4096 + n * BN;
      float mx = -3.4e38f;
      for (int m = 0; m < BN; ++m) {
        float dx = cx - coord_s[m * 3 + 0];
        float dy = cy - coord_s[m * 3 + 1];
        float dz = cz - coord_s[m * 3 + 2];
        float d  = sqrtf(dx * dx + dy * dy + dz * dz + 1e-12f);
        float s2 = 1.f / (1.f + __expf(-(d - 2.f)));
        float s4 = 1.f / (1.f + __expf(-(d - 4.f)));
        float s6 = 1.f / (1.f + __expf(-(d - 6.f)));
        float bias = dx * we[0] + dy * we[1] + dz * we[2] + d * we[3]
                   + s2 * we[4] + s4 * we[5] + s6 * we[6];
        float ok = mn * mask_s[m];
        float s = srow[m] * 0.125f + bias;     // 1/sqrt(Dh)=0.125
        s = (ok > 0.5f) ? s : -1e9f;
        srow[m] = s;
        mx = fmaxf(mx, s);
      }
      float sum = 0.f;
      for (int m = 0; m < BN; ++m) { float e = __expf(srow[m] - mx); srow[m] = e; sum += e; }
      float inv = 1.f / sum;
      __bf16* arow = attn_bf + hh * 4096 + n * BN;
      for (int m = 0; m < BN; ++m) arow[m] = (__bf16)(srow[m] * inv);
    }
    __syncthreads();

    // msg = attn @ v  -> stored bf16 into q_bf (reuse). wave: mt = w>>1, head = w&1.
    {
      const int half = lane >> 4;
      const int mt = wave >> 1;
      const int hh = wave & 1;
      v8f acc[4];
#pragma unroll
      for (int j = 0; j < 4; ++j) acc[j] = vzero();

      v16bf a = load_a(attn_bf + hh * 4096 + (mt * 16 + (lane & 15)) * BN + half * 16);
      v16bf bfr[4];
#pragma unroll
      for (int j = 0; j < 4; ++j)
        bfr[j] = *(const v16bf*)(vt_bf + ((hh * 4 + j) * 16 + (lane & 15)) * BN + half * 16);

#pragma unroll
      for (int kt = 0; kt < 2; ++kt) {
        v16bf an = a;
        v16bf bn[4];
        if (kt < 1) {
          an = load_a(attn_bf + hh * 4096 + (mt * 16 + (lane & 15)) * BN + 32 + half * 16);
#pragma unroll
          for (int j = 0; j < 4; ++j)
            bn[j] = *(const v16bf*)(vt_bf + ((hh * 4 + j) * 16 + (lane & 15)) * BN + 32 + half * 16);
        }
#pragma unroll
        for (int j = 0; j < 4; ++j)
          acc[j] = __builtin_amdgcn_wmma_f32_16x16x32_bf16(
              false, a, false, bfr[j], (short)0, acc[j], false, false);
        if (kt < 1) {
          a = an;
#pragma unroll
          for (int j = 0; j < 4; ++j) bfr[j] = bn[j];
        }
      }

      const int c = lane & 15;
      const int rbase = mt * 16 + half * 8;
#pragma unroll
      for (int j = 0; j < 4; ++j)
#pragma unroll
        for (int r = 0; r < 8; ++r)
          q_bf[(rbase + r) * FF + (hh * 4 + j) * 16 + c] = (__bf16)acc[j][r];
    }
    __syncthreads();

    // h += msg @ Wo
    gemm_a_w(q_bf, wo, h_s, nullptr, nullptr, 2, wave, lane);
    __syncthreads();

    // h += tanh(layernorm(h) @ Wf + bf)
    ln_rows_bf16(h_s, hn_bf, tid);
    __syncthreads();
    gemm_a_w(hn_bf, wf, h_s, nullptr, bf_g + l * FF, 3, wave, lane);
    __syncthreads();

    // h *= mask
    for (int idx = tid; idx < BN * FF; idx += 256) h_s[idx] *= mask_s[idx >> 7];
    __syncthreads();
  }

  // out = layernorm(h)
  if (tid < BN) {
    const float* r = h_s + tid * FF;
    float m = 0.f;
    for (int i = 0; i < FF; ++i) m += r[i];
    m *= (1.f / FF);
    float v = 0.f;
    for (int i = 0; i < FF; ++i) { float d = r[i] - m; v += d * d; }
    v *= (1.f / FF);
    float inv = rsqrtf(v + 1e-5f);
    float* o = out_g + (size_t)blockIdx.x * BN * FF + tid * FF;
    for (int i = 0; i < FF; ++i) o[i] = (r[i] - m) * inv;
  }
}

extern "C" void kernel_launch(void* const* d_in, const int* in_sizes, int n_in,
                              void* d_out, int out_size, void* d_ws, size_t ws_size,
                              hipStream_t stream) {
  (void)in_sizes; (void)n_in; (void)out_size; (void)ws_size;
  const float*         coords  = (const float*)d_in[0];
  const int*           species = (const int*)d_in[1];
  const unsigned char* mask    = (const unsigned char*)d_in[2];
  const float*         embed   = (const float*)d_in[3];
  const float*         Wq      = (const float*)d_in[4];
  const float*         Wk      = (const float*)d_in[5];
  const float*         Wv      = (const float*)d_in[6];
  const float*         Wo      = (const float*)d_in[7];
  const float*         We      = (const float*)d_in[8];
  const float*         Wf      = (const float*)d_in[9];
  const float*         bf      = (const float*)d_in[10];
  __bf16* wsw = (__bf16*)d_ws;

  convert_weights<<<(15 * MATSZ + 255) / 256, 256, 0, stream>>>(Wq, Wk, Wv, Wo, Wf, wsw);
  gnn_forward<<<2048, 256, SMEM_BYTES, stream>>>(coords, species, mask, embed, We, bf,
                                                 wsw, (float*)d_out);
}
